// GAT_15204184228309
// MI455X (gfx1250) — matrix operand order repaired and checked by reading, not compiled, not run on hardware
//
#include <hip/hip_runtime.h>
#include <hip/hip_bf16.h>

typedef __attribute__((ext_vector_type(16))) _Float16 v16h;
typedef __attribute__((ext_vector_type(8)))  _Float16 v8h;
typedef __attribute__((ext_vector_type(8)))  float    v8f;

#define N_NODES   50000
#define N_EDGES   800000
#define FD        128
#define HD        64
#define CD        10
#define GD        512
#define EN        (N_EDGES + N_NODES)
#define NEG_SLOPE 0.2f
#define BN_EPS    1e-5f
#define MINIT     (-1e30f)

// ---------------- utility kernels ----------------

__global__ void k_fill(float* __restrict__ p, float v, int n) {
  int i = blockIdx.x * blockDim.x + threadIdx.x;
  if (i < n) p[i] = v;
}

__global__ void k_f32_to_f16(const float* __restrict__ in, _Float16* __restrict__ out, int n) {
  int i = blockIdx.x * blockDim.x + threadIdx.x;
  if (i < n) out[i] = (_Float16)in[i];
}

__device__ __forceinline__ void atomicMaxF(float* addr, float v) {
  // sign-aware trick: int ordering matches float ordering for >=0,
  // reversed (as uint) for <0. Safe under mixed-sign races.
  if (v >= 0.0f) atomicMax((int*)addr, __float_as_int(v));
  else           atomicMin((unsigned int*)addr, __float_as_uint(v));
}

// ---------------- WMMA dual GEMM: outL = X*WL^T, outR = X*WR^T ----------------
// X: [rows, K] f16 row-major; WL/WR: [HD, K] f16 row-major; out: [rows, HD] f32.
// One wave per 16x16 tile; A fragment shared by both accumulators.

__global__ void k_gemm_pair(const _Float16* __restrict__ X,
                            const _Float16* __restrict__ WL,
                            const _Float16* __restrict__ WR,
                            float* __restrict__ outL,
                            float* __restrict__ outR,
                            int K)
{
  const int lane = threadIdx.x;          // 0..31 (wave32)
  const int m0   = blockIdx.x * 16;
  const int h0   = blockIdx.y * 16;
  const int l15  = lane & 15;
  const int hi16 = lane >> 4;            // 0 | 1
  const int kbA  = hi16 ? 16 : 0;        // A: lanes 16-31 hold K=16..31
  const int kbB  = hi16 ? 16 : 0;        // B: lanes 16-31 hold K=16..31

  const _Float16* xrow  = X  + (size_t)(m0 + l15) * K;
  const _Float16* wlrow = WL + (size_t)(h0 + l15) * K;
  const _Float16* wrrow = WR + (size_t)(h0 + l15) * K;

  v8f accL = {}; v8f accR = {};
  for (int k0 = 0; k0 < K; k0 += 32) {
    v8h alo = *(const v8h*)(xrow  + k0 + kbA);
    v8h ahi = *(const v8h*)(xrow  + k0 + kbA + 8);
    v8h bl0 = *(const v8h*)(wlrow + k0 + kbB);
    v8h bl1 = *(const v8h*)(wlrow + k0 + kbB + 8);
    v8h br0 = *(const v8h*)(wrrow + k0 + kbB);
    v8h br1 = *(const v8h*)(wrrow + k0 + kbB + 8);
    v16h a, bl, br;
#pragma unroll
    for (int j = 0; j < 8; ++j) {
      a[j]  = alo[j]; a[j + 8]  = ahi[j];
      bl[j] = bl0[j]; bl[j + 8] = bl1[j];
      br[j] = br0[j]; br[j + 8] = br1[j];
    }
    accL = __builtin_amdgcn_wmma_f32_16x16x32_f16(false, a, false, bl, (short)0, accL, false, false);
    accR = __builtin_amdgcn_wmma_f32_16x16x32_f16(false, a, false, br, (short)0, accR, false, false);
  }

  const int mb = m0 + (hi16 ? 8 : 0);
  const int nc = h0 + l15;
#pragma unroll
  for (int r = 0; r < 8; ++r) {
    outL[(size_t)(mb + r) * HD + nc] = accL[r];
    outR[(size_t)(mb + r) * HD + nc] = accR[r];
  }
}

// ---------------- edge kernels (wave per edge, 2 features per lane) ----------------

__global__ void k_edge_score(const int* __restrict__ ei,
                             const float* __restrict__ xl,
                             const float* __restrict__ xr,
                             const float* __restrict__ att,
                             float* __restrict__ evals,
                             float* __restrict__ mbuf)
{
  const int wid  = (blockIdx.x * blockDim.x + threadIdx.x) >> 5;
  const int lane = threadIdx.x & 31;
  if (wid >= EN) return;
  const int s = (wid < N_EDGES) ? ei[wid]            : (wid - N_EDGES);
  const int d = (wid < N_EDGES) ? ei[N_EDGES + wid]  : (wid - N_EDGES);
  const int h = lane * 2;
  float2 vl = *(const float2*)(xl + (size_t)s * HD + h);
  float2 vr = *(const float2*)(xr + (size_t)d * HD + h);
  float u0 = vl.x + vr.x, u1 = vl.y + vr.y;
  u0 = (u0 > 0.f) ? u0 : NEG_SLOPE * u0;
  u1 = (u1 > 0.f) ? u1 : NEG_SLOPE * u1;
  float p = u0 * att[h] + u1 * att[h + 1];
#pragma unroll
  for (int off = 16; off > 0; off >>= 1) p += __shfl_xor(p, off, 32);
  if (lane == 0) { evals[wid] = p; atomicMaxF(&mbuf[d], p); }
}

__global__ void k_edge_exp(const int* __restrict__ ei,
                           const float* __restrict__ evals,
                           const float* __restrict__ mbuf,
                           float* __restrict__ avals,
                           float* __restrict__ denom)
{
  int i = blockIdx.x * blockDim.x + threadIdx.x;
  if (i >= EN) return;
  int d = (i < N_EDGES) ? ei[N_EDGES + i] : (i - N_EDGES);
  float mv = mbuf[d];
  mv = (mv > -1e29f) ? mv : 0.f;  // isfinite() fallback in reference
  float av = __expf(evals[i] - mv);
  avals[i] = av;
  atomicAdd(&denom[d], av);
}

__global__ void k_edge_scatter(const int* __restrict__ ei,
                               const float* __restrict__ avals,
                               const float* __restrict__ denom,
                               const float* __restrict__ xl,
                               float* __restrict__ agg)
{
  const int wid  = (blockIdx.x * blockDim.x + threadIdx.x) >> 5;
  const int lane = threadIdx.x & 31;
  if (wid >= EN) return;
  const int s = (wid < N_EDGES) ? ei[wid]           : (wid - N_EDGES);
  const int d = (wid < N_EDGES) ? ei[N_EDGES + wid] : (wid - N_EDGES);
  const float alpha = avals[wid] / denom[d];
  const int h = lane * 2;
  float2 vl = *(const float2*)(xl + (size_t)s * HD + h);
  atomicAdd(&agg[(size_t)d * HD + h],     alpha * vl.x);
  atomicAdd(&agg[(size_t)d * HD + h + 1], alpha * vl.y);
}

// ---------------- bias/activation/pool ----------------

__global__ void k_bias_relu_h(const float* __restrict__ agg, const float* __restrict__ b,
                              _Float16* __restrict__ hh)
{
  int i = blockIdx.x * blockDim.x + threadIdx.x;
  if (i >= N_NODES * HD) return;
  float v = agg[i] + b[i & (HD - 1)];
  v = v > 0.f ? v : 0.f;
  hh[i] = (_Float16)v;
}

__global__ void k_bias2(const float* __restrict__ agg, const float* __restrict__ b,
                        float* __restrict__ h2)
{
  int i = blockIdx.x * blockDim.x + threadIdx.x;
  if (i >= N_NODES * HD) return;
  h2[i] = agg[i] + b[i & (HD - 1)];
}

__global__ void k_pool(const float* __restrict__ h2, const int* __restrict__ batch,
                       float* __restrict__ pooled, float* __restrict__ counts)
{
  int i = blockIdx.x * blockDim.x + threadIdx.x;
  if (i >= N_NODES * HD) return;
  int n = i >> 6, h = i & 63;
  int g = batch[n];
  atomicAdd(&pooled[(size_t)g * HD + h], h2[i]);
  if (h == 0) atomicAdd(&counts[g], 1.0f);
}

// ---------------- MLP head (tiny) ----------------

__global__ void k_head_linear3(const float* __restrict__ pooled, const float* __restrict__ counts,
                               const float* __restrict__ W3, const float* __restrict__ b3,
                               float* __restrict__ z)
{
  int g = blockIdx.x, ho = threadIdx.x;
  float c = counts[g]; c = c > 1.f ? c : 1.f;
  const float* pr = pooled + (size_t)g * HD;
  const float* wr = W3 + (size_t)ho * HD;
  float s = 0.f;
  for (int i = 0; i < HD; ++i) s += pr[i] * wr[i];
  z[(size_t)g * HD + ho] = s / c + b3[ho];
}

__global__ void k_bn_stats(const float* __restrict__ z, float* __restrict__ mu, float* __restrict__ var)
{
  int h = threadIdx.x;
  float s = 0.f, s2 = 0.f;
  for (int g = 0; g < GD; ++g) { float v = z[(size_t)g * HD + h]; s += v; s2 += v * v; }
  float m = s / (float)GD;
  mu[h] = m; var[h] = s2 / (float)GD - m * m;
}

__global__ void k_bn_relu(const float* __restrict__ z, const float* __restrict__ mu,
                          const float* __restrict__ var, const float* __restrict__ gamma,
                          const float* __restrict__ beta, float* __restrict__ zr)
{
  int g = blockIdx.x, h = threadIdx.x;
  float v = (z[(size_t)g * HD + h] - mu[h]) * rsqrtf(var[h] + BN_EPS) * gamma[h] + beta[h];
  zr[(size_t)g * HD + h] = v > 0.f ? v : 0.f;
}

__global__ void k_head_out(const float* __restrict__ zr, const float* __restrict__ W4,
                           const float* __restrict__ b4, float* __restrict__ out)
{
  int g = blockIdx.x * blockDim.x + threadIdx.x;
  if (g >= GD) return;
  const float* zrow = zr + (size_t)g * HD;
  float logit[CD];
  float mx = -1e30f;
#pragma unroll
  for (int c = 0; c < CD; ++c) {
    const float* wr = W4 + (size_t)c * HD;
    float s = b4[c];
    for (int i = 0; i < HD; ++i) s += zrow[i] * wr[i];
    logit[c] = s; mx = s > mx ? s : mx;
  }
  float se = 0.f;
#pragma unroll
  for (int c = 0; c < CD; ++c) se += __expf(logit[c] - mx);
  float lse = mx + logf(se);
#pragma unroll
  for (int c = 0; c < CD; ++c) out[(size_t)g * CD + c] = logit[c] - lse;
}

// ---------------- launch ----------------

extern "C" void kernel_launch(void* const* d_in, const int* in_sizes, int n_in,
                              void* d_out, int out_size, void* d_ws, size_t ws_size,
                              hipStream_t stream) {
  const float* x     = (const float*)d_in[0];
  const int*   ei    = (const int*)d_in[1];
  const int*   batch = (const int*)d_in[2];
  const float* Wl1   = (const float*)d_in[3];
  const float* Wr1   = (const float*)d_in[4];
  const float* att1  = (const float*)d_in[5];
  const float* b1    = (const float*)d_in[6];
  const float* Wl2   = (const float*)d_in[7];
  const float* Wr2   = (const float*)d_in[8];
  const float* att2  = (const float*)d_in[9];
  const float* b2    = (const float*)d_in[10];
  const float* W3    = (const float*)d_in[11];
  const float* b3    = (const float*)d_in[12];
  const float* gamma = (const float*)d_in[13];
  const float* beta  = (const float*)d_in[14];
  const float* W4    = (const float*)d_in[15];
  const float* b4    = (const float*)d_in[16];
  float* out = (float*)d_out;

  // workspace carve (256B aligned slots)
  char* p = (char*)d_ws;
  auto carve = [&](size_t bytes) -> void* {
    void* r = (void*)p;
    p += (bytes + 255) & ~(size_t)255;
    return r;
  };
  float*     xl     = (float*)carve((size_t)N_NODES * HD * 4);
  float*     xr     = (float*)carve((size_t)N_NODES * HD * 4);
  float*     agg    = (float*)carve((size_t)N_NODES * HD * 4);
  float*     h2     = (float*)carve((size_t)N_NODES * HD * 4);
  float*     mbuf   = (float*)carve((size_t)N_NODES * 4);
  float*     denom  = (float*)carve((size_t)N_NODES * 4);
  float*     evals  = (float*)carve((size_t)EN * 4);
  float*     avals  = (float*)carve((size_t)EN * 4);
  _Float16*  xh     = (_Float16*)carve((size_t)N_NODES * FD * 2);
  _Float16*  hh     = (_Float16*)carve((size_t)N_NODES * HD * 2);
  _Float16*  w1l    = (_Float16*)carve((size_t)HD * FD * 2);
  _Float16*  w1r    = (_Float16*)carve((size_t)HD * FD * 2);
  _Float16*  w2l    = (_Float16*)carve((size_t)HD * HD * 2);
  _Float16*  w2r    = (_Float16*)carve((size_t)HD * HD * 2);
  float*     pooled = (float*)carve((size_t)GD * HD * 4);
  float*     counts = (float*)carve((size_t)GD * 4);
  float*     z      = (float*)carve((size_t)GD * HD * 4);
  float*     zr     = (float*)carve((size_t)GD * HD * 4);
  float*     mu     = (float*)carve((size_t)HD * 4);
  float*     var    = (float*)carve((size_t)HD * 4);

  const int NH = N_NODES * HD;
  const int edgeWaveBlocks = (EN + 7) / 8;        // 8 waves of 32 per 256-thread block
  const int edgeThreadBlocks = (EN + 255) / 256;

  // fp16 copies of activations + weights
  k_f32_to_f16<<<(N_NODES * FD + 255) / 256, 256, 0, stream>>>(x, xh, N_NODES * FD);
  k_f32_to_f16<<<(HD * FD + 255) / 256, 256, 0, stream>>>(Wl1, w1l, HD * FD);
  k_f32_to_f16<<<(HD * FD + 255) / 256, 256, 0, stream>>>(Wr1, w1r, HD * FD);
  k_f32_to_f16<<<(HD * HD + 255) / 256, 256, 0, stream>>>(Wl2, w2l, HD * HD);
  k_f32_to_f16<<<(HD * HD + 255) / 256, 256, 0, stream>>>(Wr2, w2r, HD * HD);

  // ---- layer 1 ----
  k_fill<<<(N_NODES + 255) / 256, 256, 0, stream>>>(mbuf, MINIT, N_NODES);
  k_fill<<<(N_NODES + 255) / 256, 256, 0, stream>>>(denom, 0.f, N_NODES);
  k_fill<<<(NH + 255) / 256, 256, 0, stream>>>(agg, 0.f, NH);

  k_gemm_pair<<<dim3(N_NODES / 16, HD / 16), 32, 0, stream>>>(xh, w1l, w1r, xl, xr, FD);
  k_edge_score<<<edgeWaveBlocks, 256, 0, stream>>>(ei, xl, xr, att1, evals, mbuf);
  k_edge_exp<<<edgeThreadBlocks, 256, 0, stream>>>(ei, evals, mbuf, avals, denom);
  k_edge_scatter<<<edgeWaveBlocks, 256, 0, stream>>>(ei, avals, denom, xl, agg);
  k_bias_relu_h<<<(NH + 255) / 256, 256, 0, stream>>>(agg, b1, hh);

  // ---- layer 2 ----
  k_fill<<<(N_NODES + 255) / 256, 256, 0, stream>>>(mbuf, MINIT, N_NODES);
  k_fill<<<(N_NODES + 255) / 256, 256, 0, stream>>>(denom, 0.f, N_NODES);
  k_fill<<<(NH + 255) / 256, 256, 0, stream>>>(agg, 0.f, NH);

  k_gemm_pair<<<dim3(N_NODES / 16, HD / 16), 32, 0, stream>>>(hh, w2l, w2r, xl, xr, HD);
  k_edge_score<<<edgeWaveBlocks, 256, 0, stream>>>(ei, xl, xr, att2, evals, mbuf);
  k_edge_exp<<<edgeThreadBlocks, 256, 0, stream>>>(ei, evals, mbuf, avals, denom);
  k_edge_scatter<<<edgeWaveBlocks, 256, 0, stream>>>(ei, avals, denom, xl, agg);
  k_bias2<<<(NH + 255) / 256, 256, 0, stream>>>(agg, b2, h2);

  // ---- pool + head ----
  k_fill<<<(GD * HD + 255) / 256, 256, 0, stream>>>(pooled, 0.f, GD * HD);
  k_fill<<<(GD + 255) / 256, 256, 0, stream>>>(counts, 0.f, GD);
  k_pool<<<(NH + 255) / 256, 256, 0, stream>>>(h2, batch, pooled, counts);

  k_head_linear3<<<GD, HD, 0, stream>>>(pooled, counts, W3, b3, z);
  k_bn_stats<<<1, HD, 0, stream>>>(z, mu, var);
  k_bn_relu<<<GD, HD, 0, stream>>>(z, mu, var, gamma, beta, zr);
  k_head_out<<<(GD + 255) / 256, 256, 0, stream>>>(zr, W4, b4, out);
}
